// MultiHeadAttention_35098472743180
// MI455X (gfx1250) — matrix operand order repaired and checked by reading, not compiled, not run on hardware
//
#include <hip/hip_runtime.h>
#include <hip/hip_bf16.h>

// ---------------------------------------------------------------------------
// Problem constants (B=2, S=16384, E=256, H=2)
// ---------------------------------------------------------------------------
#define S_SEQ   16384
#define EDIM    256
#define HE      512
#define NROWS   32768        // B * S
#define HALF_S  8192         // S / H

typedef __attribute__((ext_vector_type(16))) __bf16       bf16x16;
typedef __attribute__((ext_vector_type(4)))  __bf16       bf16x4;
typedef __attribute__((ext_vector_type(8)))  float        f32x8;
typedef __attribute__((ext_vector_type(4)))  float        f32x4;
typedef __attribute__((ext_vector_type(4)))  unsigned int u32x4;

typedef __attribute__((ext_vector_type(4))) unsigned int tdm_v4u;
typedef __attribute__((ext_vector_type(8))) int          tdm_v8i;
typedef __attribute__((ext_vector_type(4))) int          tdm_v4i;

union BFrag {
    bf16x16 v;
    u32x4   u[2];
};

// ---------------------------------------------------------------------------
// Tensor Data Mover helpers (CDNA5 ISA ch.8: D# groups 0/1; groups 2/3 zero
// for <=2D tensors).  data_size=1 (2-byte bf16).  lds addr = low 32 bits of
// the generic pointer (ISA 10.2).
// ---------------------------------------------------------------------------
__device__ __forceinline__ unsigned lds_addr32(const void* p) {
    return (unsigned)(size_t)p;
}

__device__ __forceinline__ void tdm_load_2d_bf16(unsigned lds_byte_addr,
                                                 const void* gaddr,
                                                 unsigned tile_d0, unsigned tile_d1,
                                                 unsigned tensor_d0, unsigned tensor_d1,
                                                 unsigned stride0_elems) {
    const unsigned long long ga = (unsigned long long)(size_t)gaddr;
    tdm_v4u g0;
    g0[0] = 1u;                                               // count=1, user mode
    g0[1] = lds_byte_addr;                                    // lds_addr
    g0[2] = (unsigned)(ga & 0xFFFFFFFFu);                     // global_addr[31:0]
    g0[3] = (unsigned)((ga >> 32) & 0x1FFFFFFu) | (2u << 30); // addr[56:32] | type=2
    tdm_v8i g1;
    g1[0] = (int)(1u << 16);                                  // data_size=1 (2B), mask=0
    g1[1] = (int)((tensor_d0 & 0xFFFFu) << 16);               // tensor_dim0[15:0] @63:48
    g1[2] = (int)(((tensor_d0 >> 16) & 0xFFFFu) |             // tensor_dim0[31:16]
                  ((tensor_d1 & 0xFFFFu) << 16));             // tensor_dim1[15:0]
    g1[3] = (int)(((tensor_d1 >> 16) & 0xFFFFu) |             // tensor_dim1[31:16]
                  ((tile_d0 & 0xFFFFu) << 16));               // tile_dim0
    g1[4] = (int)(tile_d1 & 0xFFFFu);                         // tile_dim1 (tile_dim2=0)
    g1[5] = (int)stride0_elems;                               // tensor_dim0_stride[31:0]
    g1[6] = 0;                                                // stride0 hi / stride1 lo
    g1[7] = 0;                                                // stride1 hi
    tdm_v4i gz = {0, 0, 0, 0};
#if __clang_major__ >= 23
    tdm_v8i gz8 = {0, 0, 0, 0, 0, 0, 0, 0};
    __builtin_amdgcn_tensor_load_to_lds(g0, g1, gz, gz, gz8, 0);
#else
    __builtin_amdgcn_tensor_load_to_lds(g0, g1, gz, gz, 0);
#endif
}

// 1-D contiguous copy (tile_dim1 unused)
__device__ __forceinline__ void tdm_load_1d_bf16(unsigned lds_byte_addr,
                                                 const void* gaddr,
                                                 unsigned n_elems) {
    tdm_load_2d_bf16(lds_byte_addr, gaddr, n_elems, 0u, 0x40000000u, 1u, n_elems);
}

// ---------------------------------------------------------------------------
// Fragment loaders from LDS (CDNA5 WMMA 16x16x32 bf16 layouts, ISA 7.12.2)
//   A (16x32, MxK): lane = M (lane&15); lane<16 holds K [0,8)+[16,24),
//                   lane>=16 holds K [8,16)+[24,32)  -> two 16B chunks.
//   B (32x16, KxN): lane = N (lane&15); same K order as A (strided gather).
// ---------------------------------------------------------------------------
__device__ __forceinline__ bf16x16 lds_a_frag(const __bf16* sA, int rowInBlk,
                                              int kb, int lane) {
    const __bf16* pp = sA + rowInBlk * EDIM + kb + ((lane & 16) >> 1);
    BFrag f;
    f.u[0] = *reinterpret_cast<const u32x4*>(pp);
    f.u[1] = *reinterpret_cast<const u32x4*>(pp + 16);
    return f.v;
}

__device__ __forceinline__ bf16x16 lds_b_frag(const __bf16* sB, int kb, int n0,
                                              int lane) {
    const __bf16* pp = sB + (kb + ((lane & 16) >> 1)) * 64 + n0 + (lane & 15);   // panel stored [256 K][64 N]
    bf16x16 v;
#pragma unroll
    for (int s = 0; s < 8; ++s) {
        v[s]     = pp[s * 64];
        v[8 + s] = pp[(16 + s) * 64];
    }
    return v;
}

// ---------------------------------------------------------------------------
// 1) Convert x (fp32) -> bf16, 4 elements / thread
// ---------------------------------------------------------------------------
__global__ void __launch_bounds__(256) mha_conv_x(const float* __restrict__ x,
                                                  __bf16* __restrict__ xb) {
    const int i = (blockIdx.x * 256 + threadIdx.x) * 4;
    f32x4 f = *reinterpret_cast<const f32x4*>(x + i);
    bf16x4 b;
    b[0] = (__bf16)f[0]; b[1] = (__bf16)f[1];
    b[2] = (__bf16)f[2]; b[3] = (__bf16)f[3];
    *reinterpret_cast<bf16x4*>(xb + i) = b;
}

// ---------------------------------------------------------------------------
// 2) Convert Wk / Wq (256x512 fp32, K-major) -> bf16 (same layout)
// ---------------------------------------------------------------------------
__global__ void __launch_bounds__(256) mha_conv_w(const float* __restrict__ wk,
                                                  const float* __restrict__ wq,
                                                  __bf16* __restrict__ wbk,
                                                  __bf16* __restrict__ wbq) {
    const int i = blockIdx.x * 256 + threadIdx.x;   // 0 .. 131071
    wbk[i] = (__bf16)wk[i];
    wbq[i] = (__bf16)wq[i];
}

// ---------------------------------------------------------------------------
// 3) Fold Wv and Wo:  M[h][p][k][c] = sum_e Wv[k, p*256+e] * Wo[h*256+e, c]
//    stored bf16 as WbY[h][k*512 + (p*256+c)]  (K x N, ready for B-frags)
// ---------------------------------------------------------------------------
__global__ void __launch_bounds__(256) mha_prep_y(const float* __restrict__ Wv,
                                                  const float* __restrict__ Wo,
                                                  __bf16* __restrict__ wby) {
    const int t   = blockIdx.x * 256 + threadIdx.x; // h*131072 + k*512 + n
    const int h   = t >> 17;
    const int rem = t & 131071;
    const int k   = rem >> 9;
    const int n   = rem & 511;
    const int p   = n >> 8;
    const int c   = n & 255;
    const float* wvp = Wv + k * HE + p * 256;           // stride 1 in e
    const float* wop = Wo + (h * 256) * EDIM + c;       // stride 256 in e
    float s = 0.f;
    for (int e = 0; e < 256; ++e) s = fmaf(wvp[e], wop[e * EDIM], s);
    wby[t] = (__bf16)s;
}

// 3b) cvec[h][p][c] = sum_e bv[p*256+e] * Wo[h*256+e, c]
__global__ void __launch_bounds__(256) mha_prep_cvec(const float* __restrict__ bv,
                                                     const float* __restrict__ Wo,
                                                     float* __restrict__ cvec) {
    const int t = blockIdx.x * 256 + threadIdx.x;   // h*512 + p*256 + c
    const int h = t >> 9;
    const int p = (t >> 8) & 1;
    const int c = t & 255;
    float s = 0.f;
    for (int e = 0; e < 256; ++e)
        s = fmaf(bv[p * 256 + e], Wo[(h * 256 + e) * EDIM + c], s);
    cvec[t] = s;
}

// ---------------------------------------------------------------------------
// 4) KQ GEMM: TDM stages A tile (128x256) + K/Q weight panels (256x64 each)
//    into LDS; each wave computes a 16x64 tile of BOTH projections, keeps them
//    in registers, forms k*q diag products and writes deterministic partial
//    bins.  grid = (256 row-blocks, 8 col-blocks), 256 threads (8 waves).
//    LDS: 64KB A + 32KB Bk + 32KB Bq = 128KB.
// ---------------------------------------------------------------------------
__global__ void __launch_bounds__(256) mha_kq_gemm(const __bf16* __restrict__ Xb,
                                                   const __bf16* __restrict__ WbK,
                                                   const __bf16* __restrict__ WbQ,
                                                   const float* __restrict__ bk,
                                                   const float* __restrict__ bq,
                                                   float* __restrict__ dpart) {
    __shared__ __bf16 smem[128 * 256 + 2 * 256 * 64];
    __bf16* sA  = smem;
    __bf16* sBK = smem + 128 * 256;
    __bf16* sBQ = smem + 128 * 256 + 256 * 64;

    const int lane  = threadIdx.x & 31;
    const int wave  = threadIdx.x >> 5;
    const int cbase = blockIdx.y * 64;                  // logical col in [0,512)

    if (threadIdx.x < 32) {
        tdm_load_1d_bf16(lds_addr32(sA), Xb + (size_t)blockIdx.x * 128 * EDIM,
                         128 * EDIM);
        tdm_load_2d_bf16(lds_addr32(sBK), WbK + cbase, 64u, 256u, 512u, 256u, 512u);
        tdm_load_2d_bf16(lds_addr32(sBQ), WbQ + cbase, 64u, 256u, 512u, 256u, 512u);
        __builtin_amdgcn_s_wait_tensorcnt(0);
    }
    __syncthreads();

    f32x8 accK[4] = {};
    f32x8 accQ[4] = {};
    const int arow = wave * 16 + (lane & 15);           // row within block tile

    for (int kb = 0; kb < EDIM; kb += 32) {
        bf16x16 a = lds_a_frag(sA, arow, kb, lane);
#pragma unroll
        for (int t = 0; t < 4; ++t) {
            bf16x16 bkf = lds_b_frag(sBK, kb, t * 16, lane);
            accK[t] = __builtin_amdgcn_wmma_f32_16x16x32_bf16(
                false, a, false, bkf, (short)0, accK[t], false, false);
            bf16x16 bqf = lds_b_frag(sBQ, kb, t * 16, lane);
            accQ[t] = __builtin_amdgcn_wmma_f32_16x16x32_bf16(
                false, a, false, bqf, (short)0, accQ[t], false, false);
        }
    }

    // diag partials: prod[g] = sum over this lane's 4 cols of (k+bk)*(q+bq)
    float prod[8];
#pragma unroll
    for (int g = 0; g < 8; ++g) prod[g] = 0.f;
#pragma unroll
    for (int t = 0; t < 4; ++t) {
        const int   col = cbase + t * 16 + (lane & 15);
        const float bkv = bk[col];
        const float bqv = bq[col];
#pragma unroll
        for (int g = 0; g < 8; ++g) {
            const float kv = accK[t][g] + bkv;
            const float qv = accQ[t][g] + bqv;
            prod[g] = fmaf(kv, qv, prod[g]);
        }
    }
    // reduce across the 16 lanes of each half (rows 0-7 in lanes 0-15,
    // rows 8-15 in lanes 16-31); xor masks < 16 stay inside the half.
#pragma unroll
    for (int m = 1; m < 16; m <<= 1) {
#pragma unroll
        for (int g = 0; g < 8; ++g) prod[g] += __shfl_xor(prod[g], m, 32);
    }
    if ((lane & 15) == 0) {
        const int p    = blockIdx.y >> 2;   // 64-col block -> half p
        const int cbin = blockIdx.y & 3;    // 4 bins per half
        const int r0   = blockIdx.x * 128 + wave * 16 + ((lane >> 4) << 3);
#pragma unroll
        for (int g = 0; g < 8; ++g)
            dpart[((r0 + g) * 2 + p) * 4 + cbin] = prod[g];
    }
}

// ---------------------------------------------------------------------------
// 5) Y GEMM: Y = Xb @ M[h]  (h fixed per 128-row block; 8192 | block rows)
//    TDM stages A tile + M[h] panel into LDS.  grid = (256, 8), 256 threads.
//    LDS: 64KB A + 32KB B = 96KB.
// ---------------------------------------------------------------------------
__global__ void __launch_bounds__(256) mha_y_gemm(const __bf16* __restrict__ Xb,
                                                  const __bf16* __restrict__ WbY,
                                                  float* __restrict__ Y) {
    __shared__ __bf16 smem[128 * 256 + 256 * 64];
    __bf16* sA = smem;
    __bf16* sB = smem + 128 * 256;

    const int lane  = threadIdx.x & 31;
    const int wave  = threadIdx.x >> 5;
    const int cbase = blockIdx.y * 64;
    const int h     = (((blockIdx.x * 128) & (S_SEQ - 1)) >= HALF_S) ? 1 : 0;
    const __bf16* W = WbY + h * (EDIM * HE);

    if (threadIdx.x < 32) {
        tdm_load_1d_bf16(lds_addr32(sA), Xb + (size_t)blockIdx.x * 128 * EDIM,
                         128 * EDIM);
        tdm_load_2d_bf16(lds_addr32(sB), W + cbase, 64u, 256u, 512u, 256u, 512u);
        __builtin_amdgcn_s_wait_tensorcnt(0);
    }
    __syncthreads();

    f32x8 acc[4] = {};
    const int arow = wave * 16 + (lane & 15);

    for (int kb = 0; kb < EDIM; kb += 32) {
        bf16x16 a = lds_a_frag(sA, arow, kb, lane);
#pragma unroll
        for (int t = 0; t < 4; ++t) {
            bf16x16 bf = lds_b_frag(sB, kb, t * 16, lane);
            acc[t] = __builtin_amdgcn_wmma_f32_16x16x32_bf16(
                false, a, false, bf, (short)0, acc[t], false, false);
        }
    }
    const int row0 = blockIdx.x * 128 + wave * 16 + ((lane >> 4) << 3);
#pragma unroll
    for (int t = 0; t < 4; ++t) {
        const int col = cbase + t * 16 + (lane & 15);
#pragma unroll
        for (int g = 0; g < 8; ++g)
            Y[(row0 + g) * HE + col] = acc[t][g];
    }
}

// ---------------------------------------------------------------------------
// 6) Gating epilogue: head-softmax over (r0 = b*S + s, r1 = r0 + 8192),
//    out = a0*(Y[r0] + cvec0) + a1*(Y[r1] + cvec1) + bo.
// ---------------------------------------------------------------------------
__global__ void __launch_bounds__(256) mha_final(const float* __restrict__ Y,
                                                 const float* __restrict__ dpart,
                                                 const float* __restrict__ cvec,
                                                 const float* __restrict__ bo,
                                                 float* __restrict__ out) {
    const int orow = blockIdx.x;            // 0 .. 32767
    const int b    = orow >> 14;
    const int spp  = orow & (S_SEQ - 1);
    const int p    = spp & 1;
    const int srow = spp >> 1;
    const int r0   = b * S_SEQ + srow;      // head 0 source row
    const int r1   = r0 + HALF_S;           // head 1 source row

    float d0 = 0.f, d1 = 0.f;
#pragma unroll
    for (int i = 0; i < 4; ++i) {           // fixed-order bin sum: deterministic
        d0 += dpart[(r0 * 2 + p) * 4 + i];
        d1 += dpart[(r1 * 2 + p) * 4 + i];
    }
    const float mx = fmaxf(d0, d1);
    const float e0 = __expf(d0 - mx);
    const float e1 = __expf(d1 - mx);
    const float a0 = e0 / (e0 + e1);
    const float a1 = 1.0f - a0;

    const int c  = threadIdx.x;
    const float y0 = Y[r0 * HE + p * 256 + c] + cvec[p * 256 + c];
    const float y1 = Y[r1 * HE + p * 256 + c] + cvec[(2 + p) * 256 + c];
    out[orow * EDIM + c] = fmaf(a0, y0, fmaf(a1, y1, bo[c]));
}

// ---------------------------------------------------------------------------
// Launch
// ---------------------------------------------------------------------------
extern "C" void kernel_launch(void* const* d_in, const int* in_sizes, int n_in,
                              void* d_out, int out_size, void* d_ws, size_t ws_size,
                              hipStream_t stream) {
    (void)in_sizes; (void)n_in; (void)out_size; (void)ws_size;
    const float* x  = (const float*)d_in[0];
    const float* Wk = (const float*)d_in[1];
    const float* bk = (const float*)d_in[2];
    const float* Wq = (const float*)d_in[3];
    const float* bq = (const float*)d_in[4];
    const float* Wv = (const float*)d_in[5];
    const float* bv = (const float*)d_in[6];
    const float* Wo = (const float*)d_in[7];
    const float* bo = (const float*)d_in[8];
    float* out = (float*)d_out;

    // Workspace layout (bytes), total ~86 MB
    char* ws = (char*)d_ws;
    __bf16* Xb    = (__bf16*)(ws);                                   // 16,777,216
    __bf16* WbK   = (__bf16*)(ws + 16777216);                        //    262,144
    __bf16* WbQ   = (__bf16*)(ws + 16777216 + 262144);               //    262,144
    __bf16* WbY   = (__bf16*)(ws + 16777216 + 524288);               //    524,288
    float*  cvec  = (float*) (ws + 16777216 + 1048576);              //      4,096
    float*  dpart = (float*) (ws + 16777216 + 1048576 + 4096);       //  1,048,576
    float*  Y     = (float*) (ws + 16777216 + 2097152 + 4096);       // 67,108,864

    mha_conv_x   <<<8192, 256, 0, stream>>>(x, Xb);
    mha_conv_w   <<<512,  256, 0, stream>>>(Wk, Wq, WbK, WbQ);
    mha_prep_y   <<<1024, 256, 0, stream>>>(Wv, Wo, WbY);
    mha_prep_cvec<<<4,    256, 0, stream>>>(bv, Wo, cvec);
    mha_kq_gemm  <<<dim3(256, 8), 256, 0, stream>>>(Xb, WbK, WbQ, bk, bq, dpart);
    mha_y_gemm   <<<dim3(256, 8), 256, 0, stream>>>(Xb, WbY, Y);
    mha_final    <<<32768, 256, 0, stream>>>(Y, dpart, cvec, bo, out);
}
